// MultiHeadAttention_84662395338883
// MI455X (gfx1250) — matrix operand order repaired and checked by reading, not compiled, not run on hardware
//
#include <hip/hip_runtime.h>
#include <math.h>

// ---------- types for CDNA5 WMMA ----------
typedef __bf16 bf16;
typedef __attribute__((ext_vector_type(16))) __bf16 v16bf;
typedef __attribute__((ext_vector_type(8)))  __bf16 v8bf;
typedef __attribute__((ext_vector_type(8)))  float  v8f;

#define WMMA_BF16(a, b, c) \
  __builtin_amdgcn_wmma_f32_16x16x32_bf16(false, (a), false, (b), (short)0, (c), false, false)

// ---------- problem constants ----------
constexpr int Bsz = 4, T = 2048, DM = 1024, H = 16, Dh = 64;
constexpr int Mrows = Bsz * T;  // 8192

// ---------- f32 -> bf16 convert ----------
__global__ void cvt_f32_bf16(const float* __restrict__ in, bf16* __restrict__ out, int n) {
  int i = blockIdx.x * blockDim.x + threadIdx.x;
  int stride = gridDim.x * blockDim.x;
  for (; i < n; i += stride) out[i] = (bf16)in[i];
}

// ---------- QKV projection: [8192,1024]bf16 @ [3072,1024]^T ----------
// Scatter epilogue: Q,K -> [B,H,T,Dh] (row = token, contiguous d for QK^T fragments),
//                   V   -> [B,H,Dh,T] TRANSPOSED (contiguous t, so P@V B-fragments are b128 loads)
__global__ void qkv_gemm(const bf16* __restrict__ X, const bf16* __restrict__ W,
                         bf16* __restrict__ Q, bf16* __restrict__ Kd, bf16* __restrict__ Vt) {
  constexpr int Kdim = DM, N = 3 * DM, NT = N / 16;
  int wave = blockIdx.x * (blockDim.x >> 5) + (threadIdx.x >> 5);
  if (wave >= (Mrows / 16) * NT) return;
  int lane = threadIdx.x & 31;
  int tm = wave / NT, tn = wave % NT;
  int ln15 = lane & 15;
  int kb8 = (lane >> 4) * 8, ks16 = (lane >> 4) * 16;
  const bf16* ap = X + (size_t)(tm * 16 + ln15) * Kdim + ks16;
  const bf16* bp = W + (size_t)(tn * 16 + ln15) * Kdim + ks16;
  v8f acc = {};
  for (int k0 = 0; k0 < Kdim; k0 += 32) {
    v8bf lo = *(const v8bf*)(ap + k0);
    v8bf hi = *(const v8bf*)(ap + k0 + 8);
    v16bf a;
#pragma unroll
    for (int i = 0; i < 8; ++i) { a[i] = lo[i]; a[i + 8] = hi[i]; }
    v16bf bmat = *(const v16bf*)(bp + k0);
    __builtin_prefetch(ap + k0 + 128, 0, 3);   // WGP-scope prefetch (near caches)
    __builtin_prefetch(bp + k0 + 128, 0, 3);
    acc = WMMA_BF16(a, bmat, acc);
  }
  // epilogue: channel c -> (sel, head, d)
  int c = tn * 16 + ln15;
  int sel = c >> 10, rem = c & 1023, h = rem >> 6, d = rem & 63;
#pragma unroll
  for (int r = 0; r < 8; ++r) {
    int mo = tm * 16 + r + 8 * (lane >> 4);
    int b = mo >> 11, t = mo & (T - 1);
    size_t bh = (size_t)b * H + h;
    bf16 val = (bf16)acc[r];
    if (sel == 0)      Q [(bh * T + t) * Dh + d] = val;
    else if (sel == 1) Kd[(bh * T + t) * Dh + d] = val;
    else               Vt[(bh * Dh + d) * T + t] = val;   // transposed
  }
}

// ---------- fused attention per (b, h, 16-row q-tile) ----------
// 128 threads = 4 waves; dynamic LDS = 16*2048 floats = 128 KB (of 320 KB/WGP)
__global__ void attn_kernel(const bf16* __restrict__ Q, const bf16* __restrict__ Km,
                            const bf16* __restrict__ Vt, float* __restrict__ Wout,
                            bf16* __restrict__ Ctx) {
  extern __shared__ float smem[];  // logically [16][2048]
  int qt = blockIdx.x, h = blockIdx.y, b = blockIdx.z;
  size_t bh = (size_t)b * H + h;
  const bf16* Qb  = Q  + bh * T * Dh;
  const bf16* Kb  = Km + bh * T * Dh;
  const bf16* Vtb = Vt + bh * Dh * T;   // [Dh][T]
  int tid = threadIdx.x, wid = tid >> 5, lane = tid & 31;
  int ln15 = lane & 15;
  int kb8 = (lane >> 4) * 8, ks16 = (lane >> 4) * 16;
  int nkt = qt + 1;          // causal: only k-tiles <= q-tile
  int ncols = nkt * 16;

  // ---- phase 1: S = (Q K^T) * scale, masked, into LDS ----
  v16bf a0, a1;  // Q fragments, d-chunks [0,32) and [32,64)
  {
    const bf16* ap = Qb + (size_t)(qt * 16 + ln15) * Dh + ks16;
    v8bf l0 = *(const v8bf*)(ap),      h0 = *(const v8bf*)(ap + 8);
    v8bf l1 = *(const v8bf*)(ap + 32), h1 = *(const v8bf*)(ap + 40);
#pragma unroll
    for (int i = 0; i < 8; ++i) { a0[i] = l0[i]; a0[i + 8] = h0[i];
                                  a1[i] = l1[i]; a1[i + 8] = h1[i]; }
  }
  for (int j = wid; j < nkt; j += 4) {
    const bf16* bp = Kb + (size_t)(j * 16 + ln15) * Dh + ks16;
    v16bf b0 = *(const v16bf*)(bp);
    v16bf b1 = *(const v16bf*)(bp + 32);
    v8f acc = {};
    acc = WMMA_BF16(a0, b0, acc);
    acc = WMMA_BF16(a1, b1, acc);
#pragma unroll
    for (int r = 0; r < 8; ++r) {
      int mr = r + 8 * (lane >> 4);
      float s = acc[r] * 0.125f;               // 1/sqrt(64)
      if (j == qt && ln15 > mr) s = -INFINITY; // causal mask (diagonal tile)
      smem[mr * T + j * 16 + ln15] = s;
    }
  }
  __syncthreads();

  // ---- phase 2: row softmax in LDS; stream w to HBM (+ explicit zero tail) ----
  {
    int row = tid >> 3, sub = tid & 7;        // 8 lanes per row
    float* srow = smem + row * T;
    float mx = -INFINITY;
    for (int c = sub; c < ncols; c += 8) mx = fmaxf(mx, srow[c]);
    for (int off = 4; off; off >>= 1) mx = fmaxf(mx, __shfl_xor(mx, off, 8));
    float sum = 0.f;
    for (int c = sub; c < ncols; c += 8) { float p = __expf(srow[c] - mx); srow[c] = p; sum += p; }
    for (int off = 4; off; off >>= 1) sum += __shfl_xor(sum, off, 8);
    float inv = 1.0f / sum;
    size_t wbase = (bh * T + (size_t)(qt * 16 + row)) * T;
    for (int c = sub; c < ncols; c += 8) { float p = srow[c] * inv; srow[c] = p; Wout[wbase + c] = p; }
    int ncp = (ncols + 31) & ~31;             // pad LDS to WMMA K granularity
    for (int c = ncols + sub; c < ncp; c += 8) srow[c] = 0.f;
    for (int c = ncols + sub; c < T; c += 8) Wout[wbase + c] = 0.f;  // masked region of w
  }
  __syncthreads();

  // ---- phase 3: ctx = P @ V ; each wave owns one 16-wide slice of d_head ----
  {
    int dcol = wid * 16 + ln15;
    const bf16* vcol = Vtb + (size_t)dcol * T;  // contiguous in t
    int ncp = (ncols + 31) & ~31;
    v8f acc = {};
    for (int kk = 0; kk < ncp; kk += 32) {
      v16bf a;
#pragma unroll
      for (int i = 0; i < 8; ++i) {
        a[i]     = (bf16)smem[ln15 * T + kk + ks16 + i];
        a[i + 8] = (bf16)smem[ln15 * T + kk + ks16 + 8 + i];
      }
      v16bf bv = *(const v16bf*)(vcol + kk + ks16);   // single b128-class load
      __builtin_prefetch(vcol + kk + 128, 0, 3);
      acc = WMMA_BF16(a, bv, acc);
    }
#pragma unroll
    for (int r = 0; r < 8; ++r) {
      int t = qt * 16 + r + 8 * (lane >> 4);
      Ctx[((size_t)b * T + t) * DM + h * Dh + dcol] = (bf16)acc[r];
    }
  }
}

// ---------- output projection: [8192,1024]bf16 @ [1024,1024]^T -> f32 out ----------
__global__ void proj_gemm(const bf16* __restrict__ A, const bf16* __restrict__ W,
                          float* __restrict__ Out) {
  constexpr int Kdim = DM, N = DM, NT = N / 16;
  int wave = blockIdx.x * (blockDim.x >> 5) + (threadIdx.x >> 5);
  if (wave >= (Mrows / 16) * NT) return;
  int lane = threadIdx.x & 31;
  int tm = wave / NT, tn = wave % NT;
  int ln15 = lane & 15;
  int kb8 = (lane >> 4) * 8, ks16 = (lane >> 4) * 16;
  const bf16* ap = A + (size_t)(tm * 16 + ln15) * Kdim + ks16;
  const bf16* bp = W + (size_t)(tn * 16 + ln15) * Kdim + ks16;
  v8f acc = {};
  for (int k0 = 0; k0 < Kdim; k0 += 32) {
    v8bf lo = *(const v8bf*)(ap + k0);
    v8bf hi = *(const v8bf*)(ap + k0 + 8);
    v16bf a;
#pragma unroll
    for (int i = 0; i < 8; ++i) { a[i] = lo[i]; a[i + 8] = hi[i]; }
    v16bf bmat = *(const v16bf*)(bp + k0);
    __builtin_prefetch(ap + k0 + 128, 0, 3);
    __builtin_prefetch(bp + k0 + 128, 0, 3);
    acc = WMMA_BF16(a, bmat, acc);
  }
#pragma unroll
  for (int r = 0; r < 8; ++r) {
    int mo = tm * 16 + r + 8 * (lane >> 4);
    Out[(size_t)mo * N + tn * 16 + ln15] = acc[r];
  }
}

extern "C" void kernel_launch(void* const* d_in, const int* in_sizes, int n_in,
                              void* d_out, int out_size, void* d_ws, size_t ws_size,
                              hipStream_t stream) {
  const float* x      = (const float*)d_in[0];  // [4,2048,1024]
  const float* w_qkv  = (const float*)d_in[1];  // [3072,1024]
  const float* w_proj = (const float*)d_in[2];  // [1024,1024]
  float* out  = (float*)d_out;                      // [4,2048,1024]
  float* wout = out + (size_t)Mrows * DM;           // [4,16,2048,2048]

  // workspace carve-out (~88 MB total)
  char* p = (char*)d_ws;
  auto take = [&](size_t bytes) { char* r = p; p += (bytes + 255) & ~(size_t)255; return r; };
  bf16* xb     = (bf16*)take((size_t)Mrows * DM * 2);
  bf16* wqkvb  = (bf16*)take((size_t)3 * DM * DM * 2);
  bf16* wprojb = (bf16*)take((size_t)DM * DM * 2);
  bf16* Q      = (bf16*)take((size_t)Mrows * DM * 2);  // [B,H,T,Dh]
  bf16* K      = (bf16*)take((size_t)Mrows * DM * 2);  // [B,H,T,Dh]
  bf16* Vt     = (bf16*)take((size_t)Mrows * DM * 2);  // [B,H,Dh,T] (transposed)
  bf16* Ctx    = (bf16*)take((size_t)Mrows * DM * 2);  // [B,T,DM]

  cvt_f32_bf16<<<512, 256, 0, stream>>>(x,      xb,     Mrows * DM);
  cvt_f32_bf16<<<192, 256, 0, stream>>>(w_qkv,  wqkvb,  3 * DM * DM);
  cvt_f32_bf16<<<64,  256, 0, stream>>>(w_proj, wprojb, DM * DM);

  // QKV GEMM: (8192/16)*(3072/16) = 98304 wave-tiles, 8 waves/block
  qkv_gemm<<<(Mrows / 16) * (3 * DM / 16) / 8, 256, 0, stream>>>(xb, wqkvb, Q, K, Vt);

  // fused attention: one block per (q-tile, head, batch); 128 KB dynamic LDS
  dim3 agrid(T / 16, H, Bsz);
  attn_kernel<<<agrid, 128, 16 * T * sizeof(float), stream>>>(Q, K, Vt, wout, Ctx);

  // output projection: (8192/16)*(1024/16) = 32768 wave-tiles
  proj_gemm<<<(Mrows / 16) * (DM / 16) / 8, 256, 0, stream>>>(Ctx, wprojb, out);
}
